// ANEPrefillLayer_34119220199688
// MI455X (gfx1250) — matrix operand order repaired and checked by reading, not compiled, not run on hardware
//
#include <hip/hip_runtime.h>
#include <hip/hip_bf16.h>

typedef __bf16 bf16_t;
typedef bf16_t bf16x8  __attribute__((ext_vector_type(8)));
typedef bf16_t bf16x16 __attribute__((ext_vector_type(16)));
typedef float  f32x8   __attribute__((ext_vector_type(8)));
typedef int    i32x4_t __attribute__((ext_vector_type(4)));

#define T_TOK 4096
#define S_CTX 4096
#define HID_C 2048
#define NH    16
#define NKV   8
#define DH    128
#define INT_F 6144
#define ATT_SCALE 0.08838834764831845f  // 1/sqrt(128)

// ---------------------------------------------------------------------------
// CDNA5 async memory->LDS copy (ASYNCcnt-tracked, no VGPR staging).
// Toolchain declares the builtin with generic int4* params (probe-confirmed
// by the round-2 diagnostic); asm fallback kept for other toolchains.
// ---------------------------------------------------------------------------
static __device__ __forceinline__ void async_copy16(const bf16_t* g, bf16_t* l) {
#if __has_builtin(__builtin_amdgcn_global_load_async_to_lds_b128)
  __builtin_amdgcn_global_load_async_to_lds_b128((i32x4_t*)g, (i32x4_t*)l, 0, 0);
#else
  const unsigned loff = (unsigned)(size_t)l;  // LDS aperture: low 32 bits = offset
  asm volatile("global_load_async_to_lds_b128 %0, %1, off"
               :: "v"(loff), "v"(g)
               : "memory");
#endif
}

static __device__ __forceinline__ void wait_async0() {
#if __has_builtin(__builtin_amdgcn_s_wait_asynccnt)
  __builtin_amdgcn_s_wait_asynccnt(0);
#else
  asm volatile("s_wait_asynccnt 0x0" ::: "memory");
#endif
}

// ---------------------------------------------------------------------------
// WMMA fragment helpers (CDNA5 16x16x32 bf16 layouts, wave32)
// A (16xK=32): lane r=lane&15 -> row r; half=lane>>4 interleave {0-7,16-23}/{8-15,24-31}
// B (K=32x16): lane r -> column r; half=lane>>4 interleave {0-7,16-23}/{8-15,24-31}
// ---------------------------------------------------------------------------
static __device__ __forceinline__ f32x8 zero8() {
  f32x8 z;
#pragma unroll
  for (int i = 0; i < 8; ++i) z[i] = 0.f;
  return z;
}

static __device__ __forceinline__ bf16x16 load_a_frag(const bf16_t* base, int ld, int lane) {
  const int r = lane & 15, hf = lane >> 4;
  const bf16_t* p = base + (size_t)r * ld + hf * 8;
  union { bf16x16 v; bf16x8 q[2]; } f;
  f.q[0] = *(const bf16x8*)(p);
  f.q[1] = *(const bf16x8*)(p + 16);
  return f.v;
}

static __device__ __forceinline__ bf16x16 load_b_frag(const bf16_t* base, int ld, int lane) {
  const int r = lane & 15, hf = lane >> 4;
  const bf16_t* p = base + (size_t)r * ld + hf * 8;
  union { bf16x16 v; bf16x8 q[2]; } f;
  f.q[0] = *(const bf16x8*)(p);
  f.q[1] = *(const bf16x8*)(p + 16);
  return f.v;
}

static __device__ __forceinline__ f32x8 wmma_bf16(bf16x16 a, bf16x16 b, f32x8 c) {
  return __builtin_amdgcn_wmma_f32_16x16x32_bf16(false, a, false, b, (short)0, c,
                                                 false, false);
}

// ---------------------------------------------------------------------------
// Generic bf16 GEMM: C[M,N] = A[M,K] * B[N,K]^T   (weights stored (out,in))
// 256 threads = 8 waves; block tile 256x128; wave tile 64x64 (4x4 WMMA frags).
// A/B slabs double-buffered in LDS via GLOBAL_LOAD_ASYNC_TO_LDS_B128:
//   per K-step: A 256x32 bf16 (16KB) + B 128x32 bf16 (8KB); 1 wait + 1 barrier.
// ---------------------------------------------------------------------------
template <int OUT_BF16>
__global__ __launch_bounds__(256) void gemm_bf16_kernel(
    const bf16_t* __restrict__ A, const bf16_t* __restrict__ B,
    void* __restrict__ C, int M, int N, int K) {
  __shared__ __align__(16) bf16_t As[2][256][32];
  __shared__ __align__(16) bf16_t Bs[2][128][32];
  const int tid = threadIdx.x;
  const int lane = tid & 31, wave = tid >> 5;
  const int r = lane & 15, hf = lane >> 4;
  const int mb = blockIdx.y * 256;
  const int nb = blockIdx.x * 128;
  const int mw = (wave >> 1) * 64;  // wave tile offset in block
  const int nw = (wave & 1) * 64;
  (void)M;

  auto issue = [&](int buf, int k0) {
    // A slab: 256 rows x 32 cols = 1024 x 16B chunks, 4 per thread
#pragma unroll
    for (int t = 0; t < 4; ++t) {
      const int chunk = tid + t * 256;
      const int row = chunk >> 2, cb = chunk & 3;
      async_copy16(A + (size_t)(mb + row) * K + k0 + cb * 8, &As[buf][row][cb * 8]);
    }
    // B slab: 128 rows x 32 cols = 512 x 16B chunks, 2 per thread
#pragma unroll
    for (int t = 0; t < 2; ++t) {
      const int chunk = tid + t * 256;
      const int row = chunk >> 2, cb = chunk & 3;
      async_copy16(B + (size_t)(nb + row) * K + k0 + cb * 8, &Bs[buf][row][cb * 8]);
    }
  };

  f32x8 acc[4][4];
#pragma unroll
  for (int i = 0; i < 4; ++i)
#pragma unroll
    for (int j = 0; j < 4; ++j) acc[i][j] = zero8();

  const int nk = K >> 5;
  issue(0, 0);
  for (int kk = 0; kk < nk; ++kk) {
    const int buf = kk & 1;
    wait_async0();     // my async ops done
    __syncthreads();   // everyone's async ops done; prev buffer fully consumed
    if (kk + 1 < nk) issue(buf ^ 1, (kk + 1) << 5);

    bf16x16 af[4], bfr[4];
#pragma unroll
    for (int i = 0; i < 4; ++i) af[i] = load_a_frag(&As[buf][mw + i * 16][0], 32, lane);
#pragma unroll
    for (int j = 0; j < 4; ++j) bfr[j] = load_b_frag(&Bs[buf][nw + j * 16][0], 32, lane);
#pragma unroll
    for (int i = 0; i < 4; ++i)
#pragma unroll
      for (int j = 0; j < 4; ++j) acc[i][j] = wmma_bf16(af[i], bfr[j], acc[i][j]);
  }

#pragma unroll
  for (int i = 0; i < 4; ++i)
#pragma unroll
    for (int j = 0; j < 4; ++j)
#pragma unroll
      for (int e = 0; e < 8; ++e) {
        const size_t row = (size_t)(mb + mw + i * 16 + e + 8 * hf);
        const int col = nb + nw + j * 16 + r;
        if (OUT_BF16)
          ((bf16_t*)C)[row * N + col] = (bf16_t)acc[i][j][e];
        else
          ((float*)C)[row * N + col] = acc[i][j][e];
      }
}

// ---------------------------------------------------------------------------
// hidden_conv (1,HID,1,T) -> x (T,HID), coalesced via LDS tiles
// ---------------------------------------------------------------------------
__global__ __launch_bounds__(256) void transpose_in_kernel(
    const float* __restrict__ hc, float* __restrict__ x) {
  __shared__ float tile[32][33];
  const int t0 = blockIdx.x * 32, c0 = blockIdx.y * 32;
#pragma unroll
  for (int j = 0; j < 4; ++j) {
    int cl = threadIdx.y + j * 8;
    tile[cl][threadIdx.x] = hc[(size_t)(c0 + cl) * T_TOK + t0 + threadIdx.x];
  }
  __syncthreads();
#pragma unroll
  for (int j = 0; j < 4; ++j) {
    int tl = threadIdx.y + j * 8;
    x[(size_t)(t0 + tl) * HID_C + c0 + threadIdx.x] = tile[threadIdx.x][tl];
  }
}

// ---------------------------------------------------------------------------
// RMSNorm over HID=2048; optional residual add R; optional fp32 sum output.
// ---------------------------------------------------------------------------
__global__ __launch_bounds__(256) void rmsnorm_kernel(
    const float* __restrict__ X, const float* __restrict__ R,
    const float* __restrict__ W, float* __restrict__ SumOut,
    bf16_t* __restrict__ Yb) {
  const int t = blockIdx.x;
  const int lane = threadIdx.x & 31, wave = threadIdx.x >> 5;
  __shared__ float red[8];
  float vals[8];
  float ss = 0.f;
#pragma unroll
  for (int k = 0; k < 8; ++k) {
    int c = threadIdx.x + k * 256;
    float v = X[(size_t)t * HID_C + c];
    if (R) v += R[(size_t)t * HID_C + c];
    vals[k] = v;
    ss += v * v;
  }
#pragma unroll
  for (int m = 16; m; m >>= 1) ss += __shfl_xor(ss, m, 32);
  if (lane == 0) red[wave] = ss;
  __syncthreads();
  float tot = 0.f;
#pragma unroll
  for (int w = 0; w < 8; ++w) tot += red[w];
  const float inv = rsqrtf(tot / (float)HID_C + 1e-6f);
#pragma unroll
  for (int k = 0; k < 8; ++k) {
    int c = threadIdx.x + k * 256;
    if (SumOut) SumOut[(size_t)t * HID_C + c] = vals[k];
    Yb[(size_t)t * HID_C + c] = (bf16_t)(vals[k] * inv * W[c]);
  }
}

// ---------------------------------------------------------------------------
// Per-(t,head) RMS + RoPE for Q/K, plain copy for V.
// mode 0: Q -> bf16 [t][h*D+d];  mode 1: K -> bf16+f32 cache [h][t][d];
// mode 2: V -> bf16+f32 cache [h][t][d]
// ---------------------------------------------------------------------------
__global__ __launch_bounds__(128) void qkv_post_kernel(
    const float* __restrict__ In, const float* __restrict__ NW,
    const float* __restrict__ Cos, const float* __restrict__ Sin,
    bf16_t* __restrict__ OutB, float* __restrict__ OutF, int nh, int mode) {
  const int t = blockIdx.x, h = blockIdx.y, d = threadIdx.x;
  const int lane = threadIdx.x & 31, wave = threadIdx.x >> 5;
  __shared__ float sh[128];
  __shared__ float red[4];
  const float v = In[((size_t)t * nh + h) * DH + d];
  float outv;
  if (mode == 2) {
    outv = v;
  } else {
    float ss = v * v;
#pragma unroll
    for (int m = 16; m; m >>= 1) ss += __shfl_xor(ss, m, 32);
    if (lane == 0) red[wave] = ss;
    __syncthreads();
    const float tot = red[0] + red[1] + red[2] + red[3];
    const float inv = rsqrtf(tot / (float)DH + 1e-6f);
    const float nv = v * inv * NW[d];
    sh[d] = nv;
    __syncthreads();
    const float partner = sh[d ^ 64];
    const float rot = (d < 64) ? -partner : partner;
    outv = nv * Cos[(size_t)t * DH + d] + rot * Sin[(size_t)t * DH + d];
  }
  if (mode == 0) {
    OutB[((size_t)t * nh + h) * DH + d] = (bf16_t)outv;
  } else {
    const size_t idx = ((size_t)h * S_CTX + t) * DH + d;
    OutB[idx] = (bf16_t)outv;
    OutF[idx] = outv;  // fp32 cache output (cache fully overwritten: POS=0, T=S)
  }
}

// ---------------------------------------------------------------------------
// Flash attention: grid (T/64, H), 128 threads = 4 waves, wave = 16 q rows.
// Causal, GQA (kv head = h>>1). Online softmax with half-wave reductions.
// ---------------------------------------------------------------------------
__global__ __launch_bounds__(128) void attn_kernel(
    const bf16_t* __restrict__ Qb, const bf16_t* __restrict__ Kb,
    const bf16_t* __restrict__ Vb, bf16_t* __restrict__ Ob) {
  const int h = blockIdx.y, hk = h >> 1;
  const int lane = threadIdx.x & 31, wave = threadIdx.x >> 5;
  const int r = lane & 15, hf = lane >> 4;
  const int t0 = blockIdx.x * 64 + wave * 16;

  __shared__ __align__(16) bf16_t Vt[DH][40];      // V^T tile: [d][s], padded
  __shared__ __align__(16) bf16_t Pb[4][16][32];   // per-wave P transpose buffer

  bf16x16 qf[4];
#pragma unroll
  for (int kc = 0; kc < 4; ++kc)
    qf[kc] = load_a_frag(Qb + (size_t)t0 * (NH * DH) + h * DH + kc * 32, NH * DH, lane);

  f32x8 oacc[8];
#pragma unroll
  for (int j = 0; j < 8; ++j) oacc[j] = zero8();
  float mrow[8], lrow[8];
#pragma unroll
  for (int i = 0; i < 8; ++i) { mrow[i] = -1e30f; lrow[i] = 0.f; }

  const bf16_t* Kh = Kb + (size_t)hk * S_CTX * DH;
  const bf16_t* Vh = Vb + (size_t)hk * S_CTX * DH;
  const int nsb = blockIdx.x * 2 + 2;  // causal: s-blocks covering s <= t0+63

  for (int sb = 0; sb < nsb; ++sb) {
    const int s0 = sb * 32;
    __syncthreads();
    // stage V^T (32 x 128) into LDS as bf16, vectorized+coalesced reads
    for (int chunk = threadIdx.x; chunk < 32 * 16; chunk += 128) {
      const int s = chunk >> 4;
      const int d0 = (chunk & 15) * 8;
      bf16x8 vv = *(const bf16x8*)(Vh + (size_t)(s0 + s) * DH + d0);
#pragma unroll
      for (int e = 0; e < 8; ++e) Vt[d0 + e][s] = vv[e];
    }
    __syncthreads();

    // S = Q K^T over d (4 WMMA K-steps), two 16-wide s tiles
    f32x8 sc[2];
    sc[0] = zero8();
    sc[1] = zero8();
#pragma unroll
    for (int c = 0; c < 2; ++c)
#pragma unroll
      for (int kc = 0; kc < 4; ++kc) {
        bf16x16 kb = load_b_frag(Kh + (size_t)(s0 + c * 16) * DH + kc * 32, DH, lane);
        sc[c] = wmma_bf16(qf[kc], kb, sc[c]);
      }

    // scale + causal mask + online softmax (row = acc elem i + 8*half)
    float p0[8], p1[8], alpha[8];
#pragma unroll
    for (int i = 0; i < 8; ++i) {
      const int trow = t0 + i + 8 * hf;
      const int col0 = s0 + r, col1 = s0 + 16 + r;
      float v0 = sc[0][i] * ATT_SCALE + (col0 <= trow ? 0.f : -10000.f);
      float v1 = sc[1][i] * ATT_SCALE + (col1 <= trow ? 0.f : -10000.f);
      float mx = fmaxf(v0, v1);
#pragma unroll
      for (int m = 1; m <= 8; m <<= 1) mx = fmaxf(mx, __shfl_xor(mx, m, 32));
      const float mnew = fmaxf(mrow[i], mx);
      alpha[i] = __expf(mrow[i] - mnew);
      p0[i] = __expf(v0 - mnew);
      p1[i] = __expf(v1 - mnew);
      float rs = p0[i] + p1[i];
#pragma unroll
      for (int m = 1; m <= 8; m <<= 1) rs += __shfl_xor(rs, m, 32);
      lrow[i] = lrow[i] * alpha[i] + rs;
      mrow[i] = mnew;
    }
#pragma unroll
    for (int j = 0; j < 8; ++j)
#pragma unroll
      for (int i = 0; i < 8; ++i) oacc[j][i] *= alpha[i];

    // transpose P (C-layout) -> A-frag layout through LDS
#pragma unroll
    for (int i = 0; i < 8; ++i) {
      Pb[wave][i + 8 * hf][r] = (bf16_t)p0[i];
      Pb[wave][i + 8 * hf][16 + r] = (bf16_t)p1[i];
    }
    __syncthreads();
    const bf16x16 pf = load_a_frag(&Pb[wave][0][0], 32, lane);
#pragma unroll
    for (int j = 0; j < 8; ++j) {
      bf16x16 vf = load_b_frag(&Vt[j * 16][0], 40, lane);
      oacc[j] = wmma_bf16(pf, vf, oacc[j]);
    }
  }

#pragma unroll
  for (int j = 0; j < 8; ++j)
#pragma unroll
    for (int i = 0; i < 8; ++i) {
      const size_t t = (size_t)(t0 + i + 8 * hf);
      Ob[t * (NH * DH) + h * DH + j * 16 + r] = (bf16_t)(oacc[j][i] / lrow[i]);
    }
}

// ---------------------------------------------------------------------------
// Elementwise helpers
// ---------------------------------------------------------------------------
__global__ __launch_bounds__(256) void f32_to_bf16_kernel(
    const float* __restrict__ in, bf16_t* __restrict__ out, int n) {
  int i = blockIdx.x * 256 + threadIdx.x;
  if (i < n) out[i] = (bf16_t)in[i];
}

__global__ __launch_bounds__(256) void silu_mul_kernel(
    const bf16_t* __restrict__ G, const bf16_t* __restrict__ U,
    bf16_t* __restrict__ A, int n) {
  int i = blockIdx.x * 256 + threadIdx.x;
  if (i < n) {
    const float g = (float)G[i], u = (float)U[i];
    A[i] = (bf16_t)(g / (1.f + __expf(-g)) * u);
  }
}

// hidden + mlp, transposed to (1, HID, 1, T)
__global__ __launch_bounds__(256) void final_out_kernel(
    const float* __restrict__ Hf, const float* __restrict__ Mf,
    float* __restrict__ Out) {
  __shared__ float tile[32][33];
  const int t0 = blockIdx.x * 32, c0 = blockIdx.y * 32;
#pragma unroll
  for (int j = 0; j < 4; ++j) {
    int tl = threadIdx.y + j * 8;
    size_t idx = (size_t)(t0 + tl) * HID_C + c0 + threadIdx.x;
    tile[tl][threadIdx.x] = Hf[idx] + Mf[idx];
  }
  __syncthreads();
#pragma unroll
  for (int j = 0; j < 4; ++j) {
    int cl = threadIdx.y + j * 8;
    Out[(size_t)(c0 + cl) * T_TOK + t0 + threadIdx.x] = tile[threadIdx.x][cl];
  }
}

// ---------------------------------------------------------------------------
// Launch
// ---------------------------------------------------------------------------
extern "C" void kernel_launch(void* const* d_in, const int* in_sizes, int n_in,
                              void* d_out, int out_size, void* d_ws, size_t ws_size,
                              hipStream_t stream) {
  (void)in_sizes; (void)n_in; (void)out_size;
  const float* hc   = (const float*)d_in[0];
  const float* cosp = (const float*)d_in[1];
  const float* sinp = (const float*)d_in[2];
  // d_in[3..6]: update_mask / attn_mask / k_cache / v_cache (handled analytically)
  const float* wq  = (const float*)d_in[7];
  const float* wk  = (const float*)d_in[8];
  const float* wv  = (const float*)d_in[9];
  const float* wo  = (const float*)d_in[10];
  const float* ln1 = (const float*)d_in[11];
  const float* ln2 = (const float*)d_in[12];
  const float* qn  = (const float*)d_in[13];
  const float* kn  = (const float*)d_in[14];
  const float* wgu = (const float*)d_in[15];
  const float* wdn = (const float*)d_in[16];

  float* outp = (float*)d_out;
  float* hidden_out = outp;                                  // HID*T
  float* k_out = outp + (size_t)HID_C * T_TOK;               // HKV*S*D
  float* v_out = k_out + (size_t)NKV * S_CTX * DH;

  const size_t TH  = (size_t)T_TOK * HID_C;
  const size_t TKV = (size_t)T_TOK * NKV * DH;
  const size_t TI  = (size_t)T_TOK * INT_F;

  size_t off = 0;
  char* base = (char*)d_ws;
  auto alloc = [&](size_t bytes) -> void* {
    void* p = base + off;
    off += (bytes + 255) & ~(size_t)255;
    return p;
  };
  bf16_t* wq_b   = (bf16_t*)alloc((size_t)2048 * 2048 * 2);
  bf16_t* wk_b   = (bf16_t*)alloc((size_t)1024 * 2048 * 2);
  bf16_t* wv_b   = (bf16_t*)alloc((size_t)1024 * 2048 * 2);
  bf16_t* wo_b   = (bf16_t*)alloc((size_t)2048 * 2048 * 2);
  bf16_t* wgu_b  = (bf16_t*)alloc((size_t)2 * INT_F * 2048 * 2);
  bf16_t* wdn_b  = (bf16_t*)alloc((size_t)2048 * INT_F * 2);
  float*  x_f    = (float*)alloc(TH * 4);
  bf16_t* h1_b   = (bf16_t*)alloc(TH * 2);
  float*  q_f    = (float*)alloc(TH * 4);
  float*  k_f    = (float*)alloc(TKV * 4);
  float*  v_f    = (float*)alloc(TKV * 4);
  bf16_t* q_b    = (bf16_t*)alloc(TH * 2);
  bf16_t* k_b    = (bf16_t*)alloc(TKV * 2);
  bf16_t* v_b    = (bf16_t*)alloc(TKV * 2);
  bf16_t* o_b    = (bf16_t*)alloc(TH * 2);
  float*  attn_f = (float*)alloc(TH * 4);
  float*  hid_f  = (float*)alloc(TH * 4);
  bf16_t* h2_b   = (bf16_t*)alloc(TH * 2);
  bf16_t* gate_b = (bf16_t*)alloc(TI * 2);
  bf16_t* up_b   = (bf16_t*)alloc(TI * 2);
  bf16_t* act_b  = (bf16_t*)alloc(TI * 2);
  float*  mlp_f  = (float*)alloc(TH * 4);
  if (off > ws_size) return;  // workspace too small; nothing safe to do

  auto cv = [&](const float* src, bf16_t* dst, int n) {
    f32_to_bf16_kernel<<<(n + 255) / 256, 256, 0, stream>>>(src, dst, n);
  };
  cv(wq, wq_b, 2048 * 2048);
  cv(wk, wk_b, 1024 * 2048);
  cv(wv, wv_b, 1024 * 2048);
  cv(wo, wo_b, 2048 * 2048);
  cv(wgu, wgu_b, 2 * INT_F * 2048);
  cv(wdn, wdn_b, 2048 * INT_F);

  transpose_in_kernel<<<dim3(T_TOK / 32, HID_C / 32), dim3(32, 8), 0, stream>>>(hc, x_f);
  rmsnorm_kernel<<<T_TOK, 256, 0, stream>>>(x_f, nullptr, ln1, nullptr, h1_b);

  gemm_bf16_kernel<0><<<dim3(2048 / 128, T_TOK / 256), 256, 0, stream>>>(
      h1_b, wq_b, q_f, T_TOK, 2048, 2048);
  gemm_bf16_kernel<0><<<dim3(1024 / 128, T_TOK / 256), 256, 0, stream>>>(
      h1_b, wk_b, k_f, T_TOK, 1024, 2048);
  gemm_bf16_kernel<0><<<dim3(1024 / 128, T_TOK / 256), 256, 0, stream>>>(
      h1_b, wv_b, v_f, T_TOK, 1024, 2048);

  qkv_post_kernel<<<dim3(T_TOK, NH), 128, 0, stream>>>(q_f, qn, cosp, sinp, q_b, nullptr, NH, 0);
  qkv_post_kernel<<<dim3(T_TOK, NKV), 128, 0, stream>>>(k_f, kn, cosp, sinp, k_b, k_out, NKV, 1);
  qkv_post_kernel<<<dim3(T_TOK, NKV), 128, 0, stream>>>(v_f, kn, cosp, sinp, v_b, v_out, NKV, 2);

  attn_kernel<<<dim3(T_TOK / 64, NH), 128, 0, stream>>>(q_b, k_b, v_b, o_b);

  gemm_bf16_kernel<0><<<dim3(2048 / 128, T_TOK / 256), 256, 0, stream>>>(
      o_b, wo_b, attn_f, T_TOK, 2048, 2048);
  rmsnorm_kernel<<<T_TOK, 256, 0, stream>>>(x_f, attn_f, ln2, hid_f, h2_b);

  gemm_bf16_kernel<1><<<dim3(INT_F / 128, T_TOK / 256), 256, 0, stream>>>(
      h2_b, wgu_b, gate_b, T_TOK, INT_F, 2048);
  gemm_bf16_kernel<1><<<dim3(INT_F / 128, T_TOK / 256), 256, 0, stream>>>(
      h2_b, wgu_b + (size_t)INT_F * 2048, up_b, T_TOK, INT_F, 2048);
  silu_mul_kernel<<<(int)((TI + 255) / 256), 256, 0, stream>>>(gate_b, up_b, act_b, (int)TI);

  gemm_bf16_kernel<0><<<dim3(2048 / 128, T_TOK / 256), 256, 0, stream>>>(
      act_b, wdn_b, mlp_f, T_TOK, 2048, INT_F);
  final_out_kernel<<<dim3(T_TOK / 32, HID_C / 32), dim3(32, 8), 0, stream>>>(
      hid_f, mlp_f, hidden_out);
}
